// EquivariantProductBlock_48137993454064
// MI455X (gfx1250) — compile-verified, matrix operand order checked by reading, not run on hardware
//
#include <hip/hip_runtime.h>

// ---------------------------------------------------------------------------
// MACE EquivariantProductBlock on gfx1250 (MI455X), wave32 + WMMA bf16.
//
// Contraction collapsed to: out3[b,c,o] = sum_p x_p * s[b,c,(o,p)] with
//   s = GEMM( monomials m[b,c,0:91] , Coef[c][91 x 36] )
//   m = [1, x_q (9), x_q*x_i (81)]          (K padded 91->96, N padded 36->48)
// Coef rows: k=0 -> c1[o,p]; k=1+q -> c2[o,p,q]; k=10+9q+i -> c3[o,p,q,i].
// Then fused o3.Linear (4x [16x128]x[128x128] bf16 GEMMs) + sc skip.
//
// v3: all 9 B-fragments of a channel preloaded into registers as one load
// clause (incremental s_wait_loadcnt instead of per-WMMA full stalls);
// monomial VALU work overlaps the loads; prefetch dropped (was SYS-scope
// and forced a loop peel).
// ---------------------------------------------------------------------------

typedef __bf16 bf16_t;
typedef __attribute__((ext_vector_type(16))) __bf16 v16bf;
typedef __attribute__((ext_vector_type(8)))  float  v8f;

#define NNODES    8192
#define MULC      128
#define DIRR      9
#define NODE_TILE 16
#define NCOLS     36              // 4 outputs * 9 p
#define KMON      91              // 1 + 9 + 81 monomials
#define CFRAG_PER_CH (9 * 512)    // 3 ntiles * 3 ktiles * 32 lanes * 16 bf16
#define LINF_ELEMS   (8 * 4 * 512)
#define FSTRIDE   134             // bf16 row stride for F (bank-conflict-free)

// WMMA 16-bit A/B fragment K index for lane/slot (ISA 7.12.2):
// lane<16 : K = {kb+0..7, kb+16..23} with kb=0 ;  lane>=16 : kb=8
__device__ __forceinline__ int frag_k(int kb, int s) {
    return (s < 8) ? (kb + s) : (kb + 16 + (s - 8));
}

template<int K>
__device__ __forceinline__ float mono(const float* __restrict__ x) {
    if constexpr (K == 0)        return 1.0f;
    else if constexpr (K < 10)   return x[K - 1];
    else if constexpr (K < KMON) return x[(K - 10) / 9] * x[(K - 10) % 9];
    else                         return 0.0f;
}

// Build one 16x32 bf16 A-fragment (this lane's 16 slots) with compile-time
// monomial indices. KB = 0 (lanes 0-15) or 8 (lanes 16-31).
template<int KB, int KT>
__device__ __forceinline__ v16bf build_afrag(const float* __restrict__ x) {
    v16bf a;
#define MONO_SLOT(S) \
    a[S] = (bf16_t)mono<KT * 32 + ((S) < 8 ? (KB + (S)) : (KB + 16 + (S) - 8))>(x)
    MONO_SLOT(0);  MONO_SLOT(1);  MONO_SLOT(2);  MONO_SLOT(3);
    MONO_SLOT(4);  MONO_SLOT(5);  MONO_SLOT(6);  MONO_SLOT(7);
    MONO_SLOT(8);  MONO_SLOT(9);  MONO_SLOT(10); MONO_SLOT(11);
    MONO_SLOT(12); MONO_SLOT(13); MONO_SLOT(14); MONO_SLOT(15);
#undef MONO_SLOT
    return a;
}

template<int KB>
__device__ __forceinline__ void build_afrags(const float* __restrict__ x, v16bf a[3]) {
    a[0] = build_afrag<KB, 0>(x);
    a[1] = build_afrag<KB, 1>(x);
    a[2] = build_afrag<KB, 2>(x);
}

// ---------------------------------------------------------------------------
// Kernel 1: bake per-channel coefficient matrices (and lin weights) into
// bf16 WMMA B-fragment layout in workspace. Tiny (~4M MACs).
// ---------------------------------------------------------------------------
__global__ void __launch_bounds__(256)
precompute_coeffs(
    const float* __restrict__ U3_0, const float* __restrict__ U2_0,
    const float* __restrict__ U1_0, const float* __restrict__ U3_1,
    const float* __restrict__ U2_1, const float* __restrict__ U1_1,
    const float* __restrict__ W3_0, const float* __restrict__ W2_0,
    const float* __restrict__ W1_0, const float* __restrict__ W3_1,
    const float* __restrict__ W2_1, const float* __restrict__ W1_1,
    const float* __restrict__ lw0,  const float* __restrict__ lw1,
    bf16_t* __restrict__ cfrag, bf16_t* __restrict__ l0f,
    bf16_t* __restrict__ l1f)
{
    const int b = blockIdx.x;
    if (b < MULC) {
        const int c = b;
        for (int idx = threadIdx.x; idx < CFRAG_PER_CH; idx += blockDim.x) {
            const int frag  = idx >> 9;        // 0..8 = ntile*3 + ktile
            const int lane  = (idx >> 4) & 31;
            const int s     = idx & 15;
            const int ntile = frag / 3, ktile = frag % 3;
            const int n  = ntile * 16 + (lane & 15);
            const int kb = (lane >> 4) * 8;
            const int k  = ktile * 32 + frag_k(kb, s);
            float val = 0.0f;
            if (n < NCOLS && k < KMON) {
                const int o = n / 9, p = n % 9;
                if (k == 0) {
                    val = (o == 0) ? U1_0[p] * W1_0[c]
                                   : U1_1[(o - 1) * 9 + p] * W1_1[c];
                } else if (k <= 9) {
                    const int q = k - 1;
                    float a = 0.f;
                    if (o == 0) {
                        for (int kk = 0; kk < 3; ++kk)
                            a += U2_0[(p * 9 + q) * 3 + kk] * W2_0[kk * MULC + c];
                    } else {
                        for (int kk = 0; kk < 2; ++kk)
                            a += U2_1[(((o - 1) * 9 + p) * 9 + q) * 2 + kk] * W2_1[kk * MULC + c];
                    }
                    val = a;
                } else {
                    const int t = k - 10, q = t / 9, i = t % 9;
                    float a = 0.f;
                    if (o == 0) {
                        for (int kk = 0; kk < 7; ++kk)
                            a += U3_0[((p * 9 + q) * 9 + i) * 7 + kk] * W3_0[kk * MULC + c];
                    } else {
                        for (int kk = 0; kk < 8; ++kk)
                            a += U3_1[((((o - 1) * 9 + p) * 9 + q) * 9 + i) * 8 + kk] * W3_1[kk * MULC + c];
                    }
                    val = a;
                }
            }
            cfrag[(size_t)c * CFRAG_PER_CH + idx] = (bf16_t)val;
        }
    } else {
        const int lb = b - MULC;                       // 0..15
        const float* lw  = (lb < 8) ? lw0 : lw1;
        bf16_t*      dst = (lb < 8) ? l0f : l1f;
        const int part = lb & 7;                       // 2048 values each
        for (int j = threadIdx.x; j < 2048; j += blockDim.x) {
            const int idx  = part * 2048 + j;          // 0..16383
            const int frag = idx >> 9;                 // ntile*4 + ktile
            const int lane = (idx >> 4) & 31;
            const int s    = idx & 15;
            const int ntile = frag >> 2, ktile = frag & 3;
            const int n  = ntile * 16 + (lane & 15);
            const int kb = (lane >> 4) * 8;
            const int k  = ktile * 32 + frag_k(kb, s); // k = u (contracted), n = v
            dst[idx] = (bf16_t)lw[k * MULC + n];
        }
    }
}

// ---------------------------------------------------------------------------
// Kernel 2: fused contraction + linear + skip. 512 blocks x 256 threads.
// Each block owns 16 nodes; each of 8 waves owns 16 channels.
// ---------------------------------------------------------------------------
__global__ void __launch_bounds__(256, 2)
mace_product_kernel(
    const float* __restrict__ nf, const float* __restrict__ sc,
    const bf16_t* __restrict__ cfrag, const bf16_t* __restrict__ l0f,
    const bf16_t* __restrict__ l1f, float* __restrict__ out)
{
    __shared__ float  accbuf[8][16][49];          // per-wave s[node][outcol]
    __shared__ bf16_t Fbf[4 * 16 * FSTRIDE];      // f[(o,node)][c] bf16

    const int tid  = threadIdx.x;
    const int wave = tid >> 5;
    const int lane = tid & 31;
    const int mrow = lane & 15;               // node-in-tile / N-col selector
    const int half = lane >> 4;
    const int kb   = half * 8;                // WMMA 16-bit K striping base
    const int node_base = blockIdx.x * NODE_TILE;

    // ---------------- Stage A: symmetric contraction via WMMA ----------------
    for (int t = 0; t < 16; ++t) {
        const int c = wave + (t << 3);

        // 1) issue x loads (needed first, for the monomial build)
        const float* row = nf + (size_t)(node_base + mrow) * (MULC * DIRR);
        float x[9];
        x[0] = row[c];
        x[1] = row[MULC + 3 * c + 0];
        x[2] = row[MULC + 3 * c + 1];
        x[3] = row[MULC + 3 * c + 2];
        x[4] = row[4 * MULC + 5 * c + 0];
        x[5] = row[4 * MULC + 5 * c + 1];
        x[6] = row[4 * MULC + 5 * c + 2];
        x[7] = row[4 * MULC + 5 * c + 3];
        x[8] = row[4 * MULC + 5 * c + 4];

        // 2) issue all 9 B-fragment loads as one clause (L2-resident table);
        //    they stay in flight while the monomials are built on VALU.
        const v16bf* bbase = (const v16bf*)(cfrag + (size_t)c * CFRAG_PER_CH);
        v16bf bfr[9];
        #pragma unroll
        for (int f = 0; f < 9; ++f)
            bfr[f] = bbase[f * 32 + lane];    // frag = nt*3 + kt

        // 3) monomial A-fragments: compile-time indices per lane half
        v16bf a[3];
        if (half == 0) build_afrags<0>(x, a);
        else           build_afrags<8>(x, a);

        // 4) 9 WMMAs with incremental load waits
        v8f acc[3] = {v8f{}, v8f{}, v8f{}};
        #pragma unroll
        for (int kt = 0; kt < 3; ++kt) {
            #pragma unroll
            for (int nt = 0; nt < 3; ++nt)
                acc[nt] = __builtin_amdgcn_wmma_f32_16x16x32_bf16(
                              false, a[kt], false, bfr[nt * 3 + kt],
                              (short)0, acc[nt], false, false);
        }

        #pragma unroll
        for (int nt = 0; nt < 3; ++nt)
            #pragma unroll
            for (int r = 0; r < 8; ++r)   // D: (M = r + 8*half, N = mrow)
                accbuf[wave][r + kb][nt * 16 + mrow] = acc[nt][r];

        // Epilogue: f[b,c,o] = sum_p x_p * s[b,(o,p)]  (o = half + 2j)
        #pragma unroll
        for (int j = 0; j < 2; ++j) {
            const int o = half + 2 * j;
            float v = 0.f;
            #pragma unroll
            for (int p = 0; p < 9; ++p)
                v = fmaf(x[p], accbuf[wave][mrow][o * 9 + p], v);
            Fbf[(o * 16 + mrow) * FSTRIDE + c] = (bf16_t)v;
        }
    }
    __syncthreads();

    // ---------------- Stage B: fused o3.Linear + skip ----------------
    const float inv_sqrt_mul = 0.08838834764831845f;  // 1/sqrt(128)
    #pragma unroll
    for (int j = 0; j < 4; ++j) {
        const int task = wave * 4 + j;                // 32 tasks = 4 o x 8 ntiles
        const int o    = task >> 3;
        const int ntv  = task & 7;
        const bf16_t* lf = (o == 0) ? l0f : l1f;
        // A row for this lane: f[o][node=mrow][0:128] as bf16 pairs (words).
        const unsigned* Fw = (const unsigned*)(Fbf + (o * 16 + mrow) * FSTRIDE);
        v8f acc = {};
        #pragma unroll
        for (int kt = 0; kt < 4; ++kt) {
            union { v16bf v; unsigned u[8]; } A;
            const int w0 = (kt * 32 + kb) >> 1;       // word idx; slots 0..7
            A.u[0] = Fw[w0 + 0]; A.u[1] = Fw[w0 + 1];
            A.u[2] = Fw[w0 + 2]; A.u[3] = Fw[w0 + 3];
            A.u[4] = Fw[w0 + 8]; A.u[5] = Fw[w0 + 9]; // slots 8..15: K+16
            A.u[6] = Fw[w0 + 10]; A.u[7] = Fw[w0 + 11];
            const v16bf bv = ((const v16bf*)lf)[(ntv * 4 + kt) * 32 + lane];
            acc = __builtin_amdgcn_wmma_f32_16x16x32_bf16(
                      false, A.v, false, bv, (short)0, acc, false, false);
        }
        #pragma unroll
        for (int r = 0; r < 8; ++r) {
            const int node = node_base + r + kb;
            const int vcol = ntv * 16 + mrow;
            const int col  = (o == 0) ? vcol : (MULC + 3 * vcol + (o - 1));
            const size_t oi = (size_t)node * (4 * MULC) + col;
            out[oi] = acc[r] * inv_sqrt_mul + sc[oi];
        }
    }
}

// ---------------------------------------------------------------------------
extern "C" void kernel_launch(void* const* d_in, const int* in_sizes, int n_in,
                              void* d_out, int out_size, void* d_ws, size_t ws_size,
                              hipStream_t stream)
{
    const float* nf   = (const float*)d_in[0];
    const float* sc   = (const float*)d_in[1];
    const float* U3_0 = (const float*)d_in[2];
    const float* U2_0 = (const float*)d_in[3];
    const float* U1_0 = (const float*)d_in[4];
    const float* U3_1 = (const float*)d_in[5];
    const float* U2_1 = (const float*)d_in[6];
    const float* U1_1 = (const float*)d_in[7];
    const float* W3_0 = (const float*)d_in[8];
    const float* W2_0 = (const float*)d_in[9];
    const float* W1_0 = (const float*)d_in[10];
    const float* W3_1 = (const float*)d_in[11];
    const float* W2_1 = (const float*)d_in[12];
    const float* W1_1 = (const float*)d_in[13];
    const float* lw0  = (const float*)d_in[14];
    const float* lw1  = (const float*)d_in[15];
    float* out = (float*)d_out;

    bf16_t* cfrag = (bf16_t*)d_ws;                         // 128*4608 bf16
    bf16_t* l0f   = cfrag + (size_t)MULC * CFRAG_PER_CH;   // 16384 bf16
    bf16_t* l1f   = l0f + LINF_ELEMS;                      // 16384 bf16
    // total ws use: ~1.22 MB (L2-resident coefficient tables)

    precompute_coeffs<<<144, 256, 0, stream>>>(
        U3_0, U2_0, U1_0, U3_1, U2_1, U1_1,
        W3_0, W2_0, W1_0, W3_1, W2_1, W1_1,
        lw0, lw1, cfrag, l0f, l1f);

    mace_product_kernel<<<NNODES / NODE_TILE, 256, 0, stream>>>(
        nf, sc, cfrag, l0f, l1f, out);
}
